// DySample_11579231830245
// MI455X (gfx1250) — compile-verified, matrix operand order checked by reading, not run on hardware
//
#include <hip/hip_runtime.h>
#include <hip/hip_bf16.h>
#include <math.h>

// ---- problem constants (match setup_inputs) ----
#define B_  16
#define C_  128
#define H_  64
#define W_  64
#define G_  4
#define CG_ 32          // C_/G_
#define S_  2
#define HS_ 128         // H_*S_
#define WS_ 128
#define HW_ 4096        // H_*W_

typedef __attribute__((ext_vector_type(16))) __bf16 v16bf;
typedef __attribute__((ext_vector_type(8)))  __bf16 bf16x8;
typedef __attribute__((ext_vector_type(8)))  float  v8f;
typedef __attribute__((ext_vector_type(4)))  int    v4i;

// -----------------------------------------------------------------------------
// Converter 1: w_proj fp32 (C,C) -> bf16 (C,C). 16384 elems.
// -----------------------------------------------------------------------------
__global__ __launch_bounds__(256) void convert_w(
    const float* __restrict__ wproj, __bf16* __restrict__ wb16)
{
    const int i = blockIdx.x * 256 + threadIdx.x;   // < 16384
    wb16[i] = (__bf16)wproj[i];
}

// -----------------------------------------------------------------------------
// Converter 2: x fp32 NCHW -> bf16 pixel-major [b*HW][C] via LDS transpose.
// Block = 256 threads, tile = 32 pixels x 128 channels.
//   load phase : coalesced fp32 reads (32 consecutive w), pack 2 channels into
//                one dword, LDS tile stride 66 dwords -> conflict-free writes.
//   store phase: each thread emits 32B of contiguous bf16 via uint2 stores.
// -----------------------------------------------------------------------------
__global__ __launch_bounds__(256) void convert_x(
    const float* __restrict__ x, __bf16* __restrict__ xb16)
{
    __shared__ unsigned tile[32 * 66];              // 32 rows, 64 ch-pairs + pad
    const int t   = threadIdx.x;
    const int b   = blockIdx.x >> 7;                // HW_/32 = 128 tiles per b
    const int hw0 = (blockIdx.x & 127) << 5;        // 32-pixel tile base

    const float* __restrict__ xb = x + ((size_t)b << 19) + hw0;

#pragma unroll
    for (int it = 0; it < 8; ++it) {
        const int idx = it * 256 + t;               // < 2048
        const int n  = idx & 31;                    // pixel (fast -> coalesced)
        const int cp = idx >> 5;                    // channel pair 0..63
        const float v0 = xb[((size_t)(2 * cp)     << 12) + n];
        const float v1 = xb[((size_t)(2 * cp + 1) << 12) + n];
        union { __bf16 h[2]; unsigned u; } pk;
        pk.h[0] = (__bf16)v0; pk.h[1] = (__bf16)v1;
        tile[n * 66 + cp] = pk.u;
    }
    __syncthreads();

    // store: thread t -> pixel n = t>>3, 32B segment q = t&7 (coalesced)
    const int n = t >> 3;
    const int q = t & 7;
    unsigned* __restrict__ dst =
        (unsigned*)(xb16 + ((size_t)(b * HW_ + hw0 + n) << 7)) + q * 8;
    const unsigned* __restrict__ src = tile + n * 66 + q * 8;
#pragma unroll
    for (int j = 0; j < 4; ++j) {                   // 4 x 8B
        uint2 v = *(const uint2*)(src + 2 * j);
        *(uint2*)(dst + 2 * j) = v;
    }
}

// -----------------------------------------------------------------------------
// Kernel A: 1x1 conv (C->C GEMM) via v_wmma_f32_16x16x32_bf16, fused BN + SiLU.
// Block = 256 threads = 8 waves, 16 pixels; wave w owns co tile [16w,16w+16).
// Weight panel (128x128 bf16 = 32KB) staged in LDS via async-to-LDS.
// Fragments per CDNA5 ISA 7.12.2 (16-bit A/B):
//   lanes 0-15 : row = lane,    K in {0..7, 16..23}  (two contiguous 16B runs)
//   lanes 16-31: row = lane-16, K in {8..15, 24..31}
//   C/D f32    : VGPR r -> (M = r + 8*(lane>=16), N = lane&15)
// -----------------------------------------------------------------------------
__global__ __launch_bounds__(256) void proj_gemm_bn_silu(
    const __bf16* __restrict__ xb16,   // (B*HW, C) bf16, pixel-major
    const __bf16* __restrict__ wb16,   // (C, C) bf16
    const float* __restrict__ gamma,
    const float* __restrict__ beta,
    const float* __restrict__ mean,
    const float* __restrict__ var,
    float* __restrict__ p)             // (B,C,H,W) fp32
{
    __shared__ __bf16 wlds[C_ * C_];   // 32KB weight panel
    const int t = threadIdx.x;

    // ---- stage weights into LDS (128B per thread) ----
    {
        const __bf16* g = wb16 + t * 64;
        __bf16*       l = wlds + t * 64;
#if __has_builtin(__builtin_amdgcn_global_load_async_to_lds_b128)
#pragma unroll
        for (int i = 0; i < 8; ++i)
            __builtin_amdgcn_global_load_async_to_lds_b128(
                (__attribute__((address_space(1))) v4i*)(g + i * 8),
                (__attribute__((address_space(3))) v4i*)(l + i * 8),
                0, 0);
        asm volatile("s_wait_asynccnt 0" ::: "memory");
#else
#pragma unroll
        for (int i = 0; i < 8; ++i)
            *(bf16x8*)(l + i * 8) = *(const bf16x8*)(g + i * 8);
#endif
    }
    __syncthreads();

    const int lane   = t & 31;
    const int wave   = t >> 5;
    const int n      = lane & 15;             // pixel col / weight row
    const int half   = lane >> 4;             // K-half selector
    const int coBase = wave << 4;

    const int pix = (blockIdx.x << 4) + n;    // b*HW + hw
    const int b   = pix >> 12;
    const int hw  = pix & 4095;

    const __bf16* __restrict__ xrow = xb16 + ((size_t)pix << 7);          // 128c
    const __bf16* __restrict__ wrow = wlds + ((size_t)(coBase + n) << 7); // 128c

    v8f acc = {};
#pragma unroll
    for (int kb = 0; kb < 4; ++kb) {
        const int k0 = (kb << 5) + (half << 3);
        const bf16x8 alo = *(const bf16x8*)(wrow + k0);        // ds_load_b128
        const bf16x8 ahi = *(const bf16x8*)(wrow + k0 + 16);
        const bf16x8 blo = *(const bf16x8*)(xrow + k0);        // global b128
        const bf16x8 bhi = *(const bf16x8*)(xrow + k0 + 16);
        const v16bf va = __builtin_shufflevector(alo, ahi,
            0,1,2,3,4,5,6,7,8,9,10,11,12,13,14,15);
        const v16bf vb = __builtin_shufflevector(blo, bhi,
            0,1,2,3,4,5,6,7,8,9,10,11,12,13,14,15);
        acc = __builtin_amdgcn_wmma_f32_16x16x32_bf16(
                  false, va, false, vb, (short)0, acc, false, false);
    }

    // ---- epilogue: BN(eval) + SiLU, scatter per C/D layout ----
    const int mo = half << 3;
#pragma unroll
    for (int r = 0; r < 8; ++r) {
        const int co = coBase + r + mo;
        const float g = gamma[co] * rsqrtf(var[co] + 1e-5f);
        float v = (acc[r] - mean[co]) * g + beta[co];
        v = v / (1.0f + __expf(-v));          // SiLU
        p[((size_t)(b * C_ + co) << 12) + hw] = v;
    }
}

// -----------------------------------------------------------------------------
// Kernel B: 3x3 conv (C=128 -> 8 offset channels, pad=1) + bias + tanh.
// Block = 64 threads = one output row (b, oc, h); weights (1152 f32) in LDS.
// -----------------------------------------------------------------------------
__global__ __launch_bounds__(64) void offset_conv_tanh(
    const float* __restrict__ x,
    const float* __restrict__ w_off,
    const float* __restrict__ b_off,
    float* __restrict__ off)
{
    __shared__ float wt[C_ * 9];
    const int blk = blockIdx.x;                // b*8*64 + oc*64 + h
    const int h  = blk & 63;
    const int oc = (blk >> 6) & 7;
    const int b  = blk >> 9;

    for (int i = threadIdx.x; i < C_ * 9; i += 64)
        wt[i] = w_off[oc * (C_ * 9) + i];
    __syncthreads();

    const int w = threadIdx.x;
    float acc = b_off[oc];
    const float* __restrict__ xb = x + ((size_t)b << 19);

    for (int c = 0; c < C_; ++c) {
#pragma unroll
        for (int kh = 0; kh < 3; ++kh) {
            const int ih = h + kh - 1;
            if (ih < 0 || ih >= H_) continue;
            const float* __restrict__ row = xb + ((size_t)c << 12) + (ih << 6);
            const float* __restrict__ wk  = wt + (c * 3 + kh) * 3;
#pragma unroll
            for (int kw = 0; kw < 3; ++kw) {
                const int iw = w + kw - 1;
                if (iw >= 0 && iw < W_) acc = fmaf(row[iw], wk[kw], acc);
            }
        }
    }
    off[((size_t)(b * 8 + oc) << 12) + (h << 6) + w] = tanhf(acc);
}

// -----------------------------------------------------------------------------
// Kernel C: bilinear upsample of offsets (align_corners), build grid, then
// border-mode bilinear grid-sample of p. One thread per (b,g,ho,wo); grid math
// amortized over the 32 group channels. Stores coalesced along wo.
// -----------------------------------------------------------------------------
__global__ __launch_bounds__(256) void dysample_gather(
    const float* __restrict__ p,
    const float* __restrict__ off,
    float* __restrict__ out)
{
    const int idx = blockIdx.x * 256 + threadIdx.x;   // B*G*Hs*Ws
    const int wo = idx & 127;
    const int ho = (idx >> 7) & 127;
    const int g  = (idx >> 14) & 3;
    const int b  = idx >> 16;

    const float r = (float)(H_ - 1) / (float)(HS_ - 1);   // 63/127
    const float py = ho * r, px = wo * r;
    const int y0 = (int)py, x0 = (int)px;
    const int y1 = min(y0 + 1, H_ - 1), x1 = min(x0 + 1, W_ - 1);
    const float fy = py - (float)y0, fx = px - (float)x0;

    const float* __restrict__ oxp = off + ((size_t)(b * 8 + 2 * g) << 12);
    const float* __restrict__ oyp = oxp + HW_;

    const float w00 = (1.f - fy) * (1.f - fx), w01 = (1.f - fy) * fx;
    const float w10 = fy * (1.f - fx),         w11 = fy * fx;
    const int i00 = (y0 << 6) + x0, i01 = (y0 << 6) + x1;
    const int i10 = (y1 << 6) + x0, i11 = (y1 << 6) + x1;

    const float ox = oxp[i00]*w00 + oxp[i01]*w01 + oxp[i10]*w10 + oxp[i11]*w11;
    const float oy = oyp[i00]*w00 + oyp[i01]*w01 + oyp[i10]*w10 + oyp[i11]*w11;

    const float gx = -1.f + wo * (2.f / (WS_ - 1)) + ox * (2.f / 128.f);
    const float gy = -1.f + ho * (2.f / (HS_ - 1)) + oy * (2.f / 128.f);

    const float ixf = fminf(fmaxf((gx + 1.f) * 0.5f * (W_ - 1), 0.f), (float)(W_ - 1));
    const float iyf = fminf(fmaxf((gy + 1.f) * 0.5f * (H_ - 1), 0.f), (float)(H_ - 1));
    const int xs0 = (int)ixf, ys0 = (int)iyf;
    const int xs1 = min(xs0 + 1, W_ - 1), ys1 = min(ys0 + 1, H_ - 1);
    const float sx = ixf - (float)xs0, sy = iyf - (float)ys0;

    const float s00 = (1.f - sy) * (1.f - sx), s01 = (1.f - sy) * sx;
    const float s10 = sy * (1.f - sx),         s11 = sy * sx;
    const int j00 = (ys0 << 6) + xs0, j01 = (ys0 << 6) + xs1;
    const int j10 = (ys1 << 6) + xs0, j11 = (ys1 << 6) + xs1;

    const float* __restrict__ pb = p + ((size_t)(b * C_ + g * CG_) << 12);
    float* __restrict__ ob = out + ((size_t)(b * C_ + g * CG_) << 14) + (ho << 7) + wo;

#pragma unroll 4
    for (int cc = 0; cc < CG_; ++cc) {
        const float* __restrict__ ch = pb + ((size_t)cc << 12);
        const float v = ch[j00]*s00 + ch[j01]*s01 + ch[j10]*s10 + ch[j11]*s11;
        ob[(size_t)cc << 14] = v;
    }
}

// -----------------------------------------------------------------------------
extern "C" void kernel_launch(void* const* d_in, const int* in_sizes, int n_in,
                              void* d_out, int out_size, void* d_ws, size_t ws_size,
                              hipStream_t stream) {
    const float* x      = (const float*)d_in[0];
    const float* w_off  = (const float*)d_in[1];
    const float* b_off  = (const float*)d_in[2];
    const float* w_proj = (const float*)d_in[3];
    const float* bn_g   = (const float*)d_in[4];
    const float* bn_b   = (const float*)d_in[5];
    const float* bn_m   = (const float*)d_in[6];
    const float* bn_v   = (const float*)d_in[7];
    float* out = (float*)d_out;

    // workspace layout:
    //   p    : 8,388,608 f32  (33.55 MB)
    //   off  :   524,288 f32  ( 2.10 MB)
    //   xb16 : 8,388,608 bf16 (16.78 MB)
    //   wb16 :    16,384 bf16 ( 32 KB )
    char* ws = (char*)d_ws;
    float*  p    = (float*)ws;
    float*  off  = (float*)(ws + (size_t)B_ * C_ * HW_ * 4);
    __bf16* xb16 = (__bf16*)(ws + (size_t)B_ * C_ * HW_ * 4 + (size_t)B_ * 8 * HW_ * 4);
    __bf16* wb16 = (__bf16*)((char*)xb16 + (size_t)B_ * C_ * HW_ * 2);

    // precision/layout converters
    convert_w<<<(C_ * C_) / 256, 256, 0, stream>>>(w_proj, wb16);
    convert_x<<<B_ * (HW_ / 32), 256, 0, stream>>>(x, xb16);

    // A: WMMA GEMM (1x1 conv) + BN + SiLU. 65536 pixels / 16 per block.
    proj_gemm_bn_silu<<<(B_ * HW_) / 16, 256, 0, stream>>>(
        xb16, wb16, bn_g, bn_b, bn_m, bn_v, p);

    // B: 3x3 offset conv + tanh. One block per (b, oc, h).
    offset_conv_tanh<<<B_ * 8 * H_, 64, 0, stream>>>(x, w_off, b_off, off);

    // C: upsample offsets + grid-sample. B*G*Hs*Ws threads.
    dysample_gather<<<(B_ * G_ * HS_ * WS_) / 256, 256, 0, stream>>>(p, off, out);
}
